// TemporalAttention_42666205118829
// MI455X (gfx1250) — matrix operand order repaired and checked by reading, not compiled, not run on hardware
//
#include <hip/hip_runtime.h>

typedef __attribute__((ext_vector_type(16))) __bf16 v16bf;
typedef __attribute__((ext_vector_type(8)))  __bf16 v8bf;
typedef __attribute__((ext_vector_type(8)))  float  v8f;

#define N_CTX   32
#define DIM     512
#define NHEADS  8
#define DHEAD   64
#define QKV_C   1536   // 3 * NHEADS * DHEAD

// ---- WMMA fragment helpers per CDNA5 ISA 7.12.2 (wave32) ----

// A (16x32, 16-bit; own k order, not ISA's): lane l<16 -> row l, K = {0..15}; lane>=16 -> row, K = {16..31}
__device__ __forceinline__ v16bf frag_a(const __bf16* buf, int ld, int row0, int k0, int lane) {
  const int r  = row0 + (lane & 15);
  const int kb = k0 + ((lane & 16) ? 16 : 0);
  const __bf16* p = buf + r * ld + kb;          // byte offset multiple of 16
  v8bf lo = *(const v8bf*)(p);
  v8bf hi = *(const v8bf*)(p + 8);
  return __builtin_shufflevector(lo, hi, 0,1,2,3,4,5,6,7,8,9,10,11,12,13,14,15);
}

// B (32x16) from a PRE-PACKED buffer: (tile, lane) -> 16 contiguous bf16 (1 KiB/wave, coalesced)
__device__ __forceinline__ v16bf frag_bp(const __bf16* pk, int tile, int lane) {
  return *(const v16bf*)(pk + ((long)(tile * 32 + lane)) * 16);
}

// B = src^T : B[k][n] = src[n][k]; k contiguous per lane -> one aligned 32B LDS vector load.
__device__ __forceinline__ v16bf frag_bT(const __bf16* buf, int ld, int k0, int col0, int lane) {
  const int c  = col0 + (lane & 15);
  const int kb = k0 + ((lane & 16) ? 16 : 0);    // byte offset multiple of 32
  return *(const v16bf*)(buf + c * ld + kb);
}

// C/D (16x16 f32): VGPR i, lanes 0-15 -> row i, lanes 16-31 -> row 8+i. Row-major bf16 store.
__device__ __forceinline__ void store_c_bf16(__bf16* buf, int ld, int row0, int col0,
                                             int lane, v8f acc, float s) {
  const int r0 = row0 + ((lane & 16) ? 8 : 0);
  const int c  = col0 + (lane & 15);
#pragma unroll
  for (int i = 0; i < 8; ++i) buf[(r0 + i) * ld + c] = (__bf16)(acc[i] * s);
}

// Transposed C store into vT[vcols][32]: each lane's 8 outputs are contiguous -> one 16B LDS store.
__device__ __forceinline__ void store_c_vT(__bf16* vT, int row0, int vc0, int lane, v8f acc) {
  const int r0 = row0 + ((lane & 16) ? 8 : 0);
  const int vc = vc0 + (lane & 15);
  v8bf o;
#pragma unroll
  for (int i = 0; i < 8; ++i) o[i] = (__bf16)acc[i];
  *(v8bf*)(vT + vc * N_CTX + r0) = o;            // 16B aligned
}

// q/k -> row-major qkb (q pre-scaled by d^-0.5); v -> transposed vT
__device__ __forceinline__ void store_qkv(__bf16* qkb, __bf16* vT, int col0, int lane,
                                          v8f acc0, v8f acc1) {
  if (col0 < 2 * DIM) {
    const float s = (col0 < DIM) ? 0.125f : 1.0f;
    store_c_bf16(qkb, 1024, 0,  col0, lane, acc0, s);
    store_c_bf16(qkb, 1024, 16, col0, lane, acc1, s);
  } else {
    store_c_vT(vT, 0,  col0 - 2 * DIM, lane, acc0);
    store_c_vT(vT, 16, col0 - 2 * DIM, lane, acc1);
  }
}

// ---- kernel 0: fp32 weights -> bf16, pre-swizzled into B-fragment layout ----
// wqp: [96 ct][16 kt][32 lane][16]  from w_qkv [512][1536]
// wop: [32 ct][16 kt][32 lane][16]  from w_out [512][512]
__global__ void ta_pack_weights(const float* __restrict__ wq, const float* __restrict__ wo,
                                __bf16* __restrict__ wqp, __bf16* __restrict__ wop) {
  const int d = blockIdx.x * blockDim.x + threadIdx.x;
  const int i    = d & 15;
  const int lane = (d >> 4) & 31;
  const int kt   = (d >> 9) & 15;
  const int ct   = d >> 13;
  const int k    = kt * 32 + ((lane & 16) ? 16 : 0) + i;
  const int col  = ct * 16 + (lane & 15);
  if (d < DIM * QKV_C) wqp[d] = (__bf16)wq[k * QKV_C + col];
  if (d < DIM * DIM)   wop[d] = (__bf16)wo[k * DIM + col];
}

// ---- kernel 1: fully fused qkv-GEMM + attention + out-GEMM, one (b,m) per block ----
__global__ __launch_bounds__(256, 1) void ta_fused(
    const float* __restrict__ x, const float* __restrict__ pos_bias,
    const __bf16* __restrict__ wqp, const __bf16* __restrict__ wop,
    float* __restrict__ out)
{
  // 160 KiB LDS pool:
  //   [0,      32768)  xb  [32][512] bf16    -- overlaid by ob after phase 2
  //   [32768,  65536)  sim [8][32][32] f32   -- attn bf16 overlays bytes 2048.. per head block
  //   [65536, 131072)  qk  [32][1024] bf16   -- q (cols 0..511, pre-scaled) and k (512..1023)
  //   [131072,163840)  vT  [512][32] bf16    -- v stored transposed
  __shared__ __align__(64) char smem[163840];
  __bf16* xb   = (__bf16*)smem;
  __bf16* ob   = (__bf16*)smem;
  float*  simp = (float*)(smem + 32768);
  __bf16* qkb  = (__bf16*)(smem + 65536);
  __bf16* vT   = (__bf16*)(smem + 131072);

  const int tid  = threadIdx.x;
  const int lane = tid & 31;
  const int wave = tid >> 5;
  const long g   = blockIdx.x;            // (b*m) index

  // ---- phase 1: load x block (vectorized), convert to bf16 ----
  const float* xg = x + g * (long)(N_CTX * DIM);
  const float4* xg4 = (const float4*)xg;
  for (int i = tid; i < (N_CTX * DIM) / 8; i += 256) {
    float4 f0 = xg4[2 * i], f1 = xg4[2 * i + 1];
    v8bf o;
    o[0] = (__bf16)f0.x; o[1] = (__bf16)f0.y; o[2] = (__bf16)f0.z; o[3] = (__bf16)f0.w;
    o[4] = (__bf16)f1.x; o[5] = (__bf16)f1.y; o[6] = (__bf16)f1.z; o[7] = (__bf16)f1.w;
    *(v8bf*)(xb + i * 8) = o;
  }
  __syncthreads();

  // ---- phase 2: qkv = xb @ w_qkv ; 2x2 blocking: 2 column tiles x 2 row tiles per wave ----
  for (int ct = wave; ct < 96; ct += 16) {       // pairs (ct, ct+8); 6 iters/wave, K = 512
    const int ctB = ct + 8;
    v8f acc0a = {}, acc1a = {}, acc0b = {}, acc1b = {};
#pragma unroll 2
    for (int kt = 0; kt < 16; ++kt) {
      v16bf ba = frag_bp(wqp, ct  * 16 + kt, lane);
      v16bf bb = frag_bp(wqp, ctB * 16 + kt, lane);
      v16bf a0 = frag_a(xb, DIM, 0,  kt * 32, lane);
      v16bf a1 = frag_a(xb, DIM, 16, kt * 32, lane);
      acc0a = __builtin_amdgcn_wmma_f32_16x16x32_bf16(false, a0, false, ba, (short)0, acc0a, false, false);
      acc1a = __builtin_amdgcn_wmma_f32_16x16x32_bf16(false, a1, false, ba, (short)0, acc1a, false, false);
      acc0b = __builtin_amdgcn_wmma_f32_16x16x32_bf16(false, a0, false, bb, (short)0, acc0b, false, false);
      acc1b = __builtin_amdgcn_wmma_f32_16x16x32_bf16(false, a1, false, bb, (short)0, acc1b, false, false);
    }
    store_qkv(qkb, vT, ct  * 16, lane, acc0a, acc1a);
    store_qkv(qkb, vT, ctB * 16, lane, acc0b, acc1b);
  }
  __syncthreads();

  // ---- phase 3: attention, one head per wave ----
  {
    const int h = wave;
    const __bf16* qh  = qkb + h * DHEAD;               // [32][.], ld = 1024
    const __bf16* kh  = qkb + DIM + h * DHEAD;
    const __bf16* vTh = vT + h * DHEAD * N_CTX;        // [64][32]
    float* simw = simp + h * (N_CTX * N_CTX);
    const float* pb = pos_bias + h * (N_CTX * N_CTX);

    // sim = q @ k^T + pos_bias   (K = 64 over head dim)
#pragma unroll
    for (int it = 0; it < 2; ++it)
#pragma unroll
      for (int jt = 0; jt < 2; ++jt) {
        v8f acc = {};
#pragma unroll
        for (int kt = 0; kt < 2; ++kt) {
          v16bf a = frag_a(qh, 1024, it * 16, kt * 32, lane);
          v16bf b = frag_bT(kh, 1024, kt * 32, jt * 16, lane);
          acc = __builtin_amdgcn_wmma_f32_16x16x32_bf16(false, a, false, b, (short)0, acc, false, false);
        }
        const int r0 = it * 16 + ((lane & 16) ? 8 : 0);
        const int c  = jt * 16 + (lane & 15);
#pragma unroll
        for (int i = 0; i < 8; ++i)
          simw[(r0 + i) * N_CTX + c] = acc[i] + pb[(r0 + i) * N_CTX + c];
      }

    // softmax: wave32 lane == row (n == 32). Row pulled into regs before attn overlay write.
    float rowv[N_CTX];
#pragma unroll
    for (int j = 0; j < N_CTX; ++j) rowv[j] = simw[lane * N_CTX + j];
    float mx = rowv[0];
#pragma unroll
    for (int j = 1; j < N_CTX; ++j) mx = fmaxf(mx, rowv[j]);
    float sum = 0.f;
#pragma unroll
    for (int j = 0; j < N_CTX; ++j) { rowv[j] = __expf(rowv[j] - mx); sum += rowv[j]; }
    const float inv = 1.f / sum;
    __bf16* attnw = (__bf16*)((char*)simw + 2048);     // overlays sim rows 16..31 (already in regs)
#pragma unroll
    for (int j = 0; j < N_CTX; ++j) attnw[lane * N_CTX + j] = (__bf16)(rowv[j] * inv);

    // out_h = attn @ v   (K = 32; V fragments contiguous from vT)
#pragma unroll
    for (int it = 0; it < 2; ++it) {
      v16bf a = frag_a(attnw, N_CTX, it * 16, 0, lane);
#pragma unroll
      for (int dt = 0; dt < 4; ++dt) {
        v16bf b = frag_bT(vTh, N_CTX, 0, dt * 16, lane);
        v8f acc = {};
        acc = __builtin_amdgcn_wmma_f32_16x16x32_bf16(false, a, false, b, (short)0, acc, false, false);
        store_c_bf16(ob, DIM, it * 16, h * DHEAD + dt * 16, lane, acc, 1.0f);
      }
    }
  }
  __syncthreads();

  // ---- phase 4: out = ob @ w_out, f32 straight to HBM; 2x2 blocking ----
  float* og = out + g * (long)(N_CTX * DIM);
  for (int ct = wave; ct < 32; ct += 16) {       // pairs (ct, ct+8); 2 iters/wave, K = 512
    const int ctB = ct + 8;
    v8f acc0a = {}, acc1a = {}, acc0b = {}, acc1b = {};
#pragma unroll 2
    for (int kt = 0; kt < 16; ++kt) {
      v16bf ba = frag_bp(wop, ct  * 16 + kt, lane);
      v16bf bb = frag_bp(wop, ctB * 16 + kt, lane);
      v16bf a0 = frag_a(ob, DIM, 0,  kt * 32, lane);
      v16bf a1 = frag_a(ob, DIM, 16, kt * 32, lane);
      acc0a = __builtin_amdgcn_wmma_f32_16x16x32_bf16(false, a0, false, ba, (short)0, acc0a, false, false);
      acc1a = __builtin_amdgcn_wmma_f32_16x16x32_bf16(false, a1, false, ba, (short)0, acc1a, false, false);
      acc0b = __builtin_amdgcn_wmma_f32_16x16x32_bf16(false, a0, false, bb, (short)0, acc0b, false, false);
      acc1b = __builtin_amdgcn_wmma_f32_16x16x32_bf16(false, a1, false, bb, (short)0, acc1b, false, false);
    }
    const int r0 = (lane & 16) ? 8 : 0;
    const int ca = ct  * 16 + (lane & 15);
    const int cb = ctB * 16 + (lane & 15);
#pragma unroll
    for (int i = 0; i < 8; ++i) og[(r0 + i) * DIM + ca]      = acc0a[i];
#pragma unroll
    for (int i = 0; i < 8; ++i) og[(16 + r0 + i) * DIM + ca] = acc1a[i];
#pragma unroll
    for (int i = 0; i < 8; ++i) og[(r0 + i) * DIM + cb]      = acc0b[i];
#pragma unroll
    for (int i = 0; i < 8; ++i) og[(16 + r0 + i) * DIM + cb] = acc1b[i];
  }
}

extern "C" void kernel_launch(void* const* d_in, const int* in_sizes, int n_in,
                              void* d_out, int out_size, void* d_ws, size_t ws_size,
                              hipStream_t stream) {
  (void)n_in; (void)out_size; (void)ws_size;
  const float* x  = (const float*)d_in[0];
  const float* pb = (const float*)d_in[1];
  const float* wq = (const float*)d_in[2];
  const float* wo = (const float*)d_in[3];
  float* out = (float*)d_out;

  __bf16* wqp = (__bf16*)d_ws;                               // 512*1536 bf16 packed
  __bf16* wop = (__bf16*)((char*)d_ws + DIM * QKV_C * 2);    // 512*512  bf16 packed (~2 MB ws)

  const int G = in_sizes[0] / (N_CTX * DIM);                 // b*m = 2048 blocks

  ta_pack_weights<<<(DIM * QKV_C + 255) / 256, 256, 0, stream>>>(wq, wo, wqp, wop);
  ta_fused<<<G, 256, 0, stream>>>(x, pb, wqp, wop, out);
}
